// WindowedMaskedAttentionBlock_55387898249975
// MI455X (gfx1250) — matrix operand order repaired and checked by reading, not compiled, not run on hardware
//
#include <hip/hip_runtime.h>
#include <hip/hip_bf16.h>
#include <stdint.h>

typedef __bf16 v8bf  __attribute__((ext_vector_type(8)));
typedef __bf16 v16bf __attribute__((ext_vector_type(16)));
typedef float  v8f   __attribute__((ext_vector_type(8)));

#define B_  2
#define P_  64
#define N_  256
#define C_  512
#define H_  8
#define HD_ 64
#define MLP_ 2048
#define M_TOT (B_ * P_ * N_)   // 32768 rows for all flattened GEMMs

// ---------------------------------------------------------------------------
// helpers
// ---------------------------------------------------------------------------
__device__ inline v8f zero8() {
    v8f z; for (int i = 0; i < 8; ++i) z[i] = 0.0f; return z;
}

typedef __attribute__((address_space(3))) char lds_char;

// workgroup-relative LDS byte offset of a __shared__ pointer
__device__ inline uint32_t lds_addr_of(const void* p) {
    return (uint32_t)(uintptr_t)(lds_char*)p;
}

// gfx1250 async DMA: global -> LDS, 16 bytes per lane, tracked by ASYNCcnt
__device__ inline void async_load_b128(uint32_t lds_off, const void* gaddr) {
    asm volatile("global_load_async_to_lds_b128 %0, %1, off"
                 :: "v"(lds_off), "v"((uint64_t)(uintptr_t)gaddr)
                 : "memory");
}
__device__ inline void wait_async0() {
    asm volatile("s_wait_asynccnt 0x0" ::: "memory");
}

// A-operand fragment (16x32 bf16): lane<16 -> row=lane, K = k0+[0..7] and k0+[16..23]
//                                  lane>=16 -> row=lane-16, K = k0+[8..15] and k0+[24..31]
__device__ inline v16bf frag_ld_a(const __bf16* base, int row, int stride, int k0, int lane) {
    int h8 = ((lane >> 4) & 1) * 8;
    const __bf16* p = base + row * stride + k0 + h8;
    union { v16bf v; v8bf h[2]; } u;
    u.h[0] = *(const v8bf*)(p);
    u.h[1] = *(const v8bf*)(p + 16);
    return u.v;
}

// B-operand fragment (32x16 bf16): lane<16 -> col=lane, K = k0+[0..7] and k0+[16..23]
//                                  lane>=16 -> col=lane-16, K = k0+[8..15] and k0+[24..31]
__device__ inline v16bf frag_ld_b(const __bf16* base, int col, int stride, int k0, int lane) {
    int h16 = ((lane >> 4) & 1) * 8;
    const __bf16* p = base + col * stride + k0 + h16;
    union { v16bf v; v8bf h[2]; } u;
    u.h[0] = *(const v8bf*)(p);
    u.h[1] = *(const v8bf*)(p + 16);
    return u.v;
}

__device__ inline v8f wmma_bf16(v16bf a, v16bf b, v8f c) {
    return __builtin_amdgcn_wmma_f32_16x16x32_bf16(
        /*neg_a=*/false, a, /*neg_b=*/false, b,
        /*c_mod=*/(short)0, c, /*reuse_a=*/false, /*reuse_b=*/false);
}

__device__ inline float gelu_exact(float v) {
    return 0.5f * v * (1.0f + erff(v * 0.70710678118654752f));
}

// ---------------------------------------------------------------------------
// fp32 -> bf16 conversion (weights)
// ---------------------------------------------------------------------------
__global__ __launch_bounds__(256) void f32_to_bf16_kernel(const float* __restrict__ s,
                                                          __bf16* __restrict__ d, int n) {
    int i = blockIdx.x * 256 + threadIdx.x;
    if (i < n) d[i] = (__bf16)s[i];
}

// ---------------------------------------------------------------------------
// Group norm over the N axis (axis=2 of (B,P,N,C)), per (b,p,c) column.
// One block per (b,p); 256 threads each own 2 channels. Output bf16.
// ---------------------------------------------------------------------------
__global__ __launch_bounds__(256) void groupnorm_kernel(const float* __restrict__ x,
                                                        const float* __restrict__ w,
                                                        const float* __restrict__ b,
                                                        __bf16* __restrict__ out) {
    int bp = blockIdx.x;
    const float* xp = x + (size_t)bp * N_ * C_;
    __bf16* op = out + (size_t)bp * N_ * C_;
    for (int c = threadIdx.x; c < C_; c += 256) {
        float s = 0.f, ss = 0.f;
        for (int n = 0; n < N_; ++n) {
            float v = xp[n * C_ + c];
            s += v; ss += v * v;
        }
        float mean = s * (1.0f / N_);
        float var  = ss * (1.0f / N_) - mean * mean;
        float inv  = rsqrtf(var + 1e-5f);
        float sw = w[c] * inv;
        float sb = b[c] - mean * sw;
        for (int n = 0; n < N_; ++n)
            op[n * C_ + c] = (__bf16)(xp[n * C_ + c] * sw + sb);
    }
}

// ---------------------------------------------------------------------------
// bf16 WMMA GEMM: out[m,n] = epi( sum_k A[m,k]*W[n,k] + bias[n] )
// 128x128 block tile, BK=32, 8 waves (4x2), each wave 32x64 (2x4 WMMA tiles).
// Double-buffered LDS filled with gfx1250 async global->LDS DMA.
// EPI: 0 = bias, 1 = bias+GELU, 2 = bias+residual(fp32)
// OBF: output dtype bf16 (true) or fp32 (false)
// ---------------------------------------------------------------------------
template<int EPI, bool OBF>
__global__ __launch_bounds__(256) void gemm_bf16_kernel(const __bf16* __restrict__ A,
                                                        const __bf16* __restrict__ W,
                                                        const float* __restrict__ bias,
                                                        const float* __restrict__ res,
                                                        void* __restrict__ outp,
                                                        int N, int K) {
    __shared__ __bf16 aT[2][128 * 40];   // padded stride 40 (80B) to spread banks
    __shared__ __bf16 bT[2][128 * 40];

    int tid  = threadIdx.x;
    int lane = tid & 31;
    int wave = tid >> 5;
    int wm = wave & 3;        // 4 row groups of 32
    int wn = wave >> 2;       // 2 col groups of 64
    int m0 = blockIdx.y * 128;
    int n0 = blockIdx.x * 128;

    v8f acc[2][4];
    for (int i = 0; i < 2; ++i)
        for (int j = 0; j < 4; ++j) acc[i][j] = zero8();

    int lr = tid >> 1;                 // 0..127
    int lseg = (tid & 1) * 16;         // 0 or 16 elems

    const uint32_t a_lds0 = lds_addr_of(&aT[0][lr * 40 + lseg]);
    const uint32_t b_lds0 = lds_addr_of(&bT[0][lr * 40 + lseg]);
    const uint32_t bufstride = 128 * 40 * 2;   // bytes between the two buffers
    const char* gA = (const char*)(A + (size_t)(m0 + lr) * K + lseg);
    const char* gW = (const char*)(W + (size_t)(n0 + lr) * K + lseg);

    // prologue: fill buffer 0 for kk = 0
    {
        async_load_b128(a_lds0,      gA);
        async_load_b128(a_lds0 + 16, gA + 16);
        async_load_b128(b_lds0,      gW);
        async_load_b128(b_lds0 + 16, gW + 16);
    }
    wait_async0();
    __syncthreads();

    int buf = 0;
    for (int kk = 0; kk < K; kk += 32) {
        // prefetch next K-slice into the other buffer (async, overlapped with WMMA)
        if (kk + 32 < K) {
            uint32_t ao = a_lds0 + (buf ^ 1) * bufstride;
            uint32_t bo = b_lds0 + (buf ^ 1) * bufstride;
            const char* pa = gA + (size_t)(kk + 32) * 2;
            const char* pw = gW + (size_t)(kk + 32) * 2;
            async_load_b128(ao,      pa);
            async_load_b128(ao + 16, pa + 16);
            async_load_b128(bo,      pw);
            async_load_b128(bo + 16, pw + 16);
        }

        v16bf af[2], bfg[4];
        for (int i = 0; i < 2; ++i)
            af[i] = frag_ld_a(aT[buf], wm * 32 + i * 16 + (lane & 15), 40, 0, lane);
        for (int j = 0; j < 4; ++j)
            bfg[j] = frag_ld_b(bT[buf], wn * 64 + j * 16 + (lane & 15), 40, 0, lane);
        for (int i = 0; i < 2; ++i)
            for (int j = 0; j < 4; ++j)
                acc[i][j] = wmma_bf16(af[i], bfg[j], acc[i][j]);

        wait_async0();
        __syncthreads();
        buf ^= 1;
    }

    int half8 = ((lane >> 4) & 1) * 8;
    for (int i = 0; i < 2; ++i) {
        for (int j = 0; j < 4; ++j) {
            int rb = m0 + wm * 32 + i * 16;
            int cb = n0 + wn * 64 + j * 16 + (lane & 15);
            float bv = bias[cb];
            for (int r = 0; r < 8; ++r) {
                int row = rb + r + half8;
                float v = acc[i][j][r] + bv;
                if (EPI == 1) v = gelu_exact(v);
                if (EPI == 2) v += res[(size_t)row * N + cb];
                if (OBF) ((__bf16*)outp)[(size_t)row * N + cb] = (__bf16)v;
                else     ((float*) outp)[(size_t)row * N + cb] = v;
            }
        }
    }
}

// ---------------------------------------------------------------------------
// Attention: one block per (b,p,h, 64-row strip). 256 threads / 8 waves.
// scores = (Q K^T)*scale + pos, masked, softmax, out = P V.  Dynamic LDS.
// K/Q tiles staged with async global->LDS DMA; V transposed manually.
// ---------------------------------------------------------------------------
#define ATTN_SMEM_BYTES ((256*72 + 64*264 + 64*72) * 2 + (64*256 + 512) * 4 + 64*264*2)

__global__ __launch_bounds__(256) void attn_kernel(const __bf16* __restrict__ qkv,
                                                   const float* __restrict__ pos,
                                                   const int* __restrict__ mask,
                                                   __bf16* __restrict__ outa) {
    extern __shared__ char smem[];
    __bf16* kT = (__bf16*)smem;                    // 256 x 72 (row=token, k-contig)
    __bf16* vT = kT + 256 * 72;                    // 64 x 264 (row=d,   k=token contig)
    __bf16* qT = vT + 64 * 264;                    // 64 x 72  (row=query row)
    float*  sS = (float*)(qT + 64 * 72);           // 64 x 256 fp32 scores
    float*  red = sS + 64 * 256;                   // 512 floats reduction scratch
    __bf16* pT = (__bf16*)(red + 512);             // 64 x 264 bf16 probs

    int rowblk = blockIdx.x;   // 0..3 (strips of 64 query rows)
    int h      = blockIdx.y;   // head
    int bp     = blockIdx.z;   // b*P + p
    int tid = threadIdx.x, lane = tid & 31, wave = tid >> 5;
    size_t rowbase = (size_t)bp * N_;

    // --- stage K (async), Q (async), V (manual transpose) into LDS ---
    {
        int n = tid;  // one token per thread
        uint32_t kdst = lds_addr_of(kT + n * 72);
        const char* ks = (const char*)(qkv + (rowbase + n) * (3 * C_) + C_ + h * HD_);
        for (int j = 0; j < 8; ++j)
            async_load_b128(kdst + j * 16, ks + j * 16);

        int r = tid >> 2, seg = (tid & 3) * 16;
        uint32_t qdst = lds_addr_of(qT + r * 72 + seg);
        const char* qs = (const char*)(qkv + (rowbase + rowblk * 64 + r) * (3 * C_) + h * HD_ + seg);
        async_load_b128(qdst,      qs);
        async_load_b128(qdst + 16, qs + 16);

        const __bf16* vs = qkv + (rowbase + n) * (3 * C_) + 2 * C_ + h * HD_;
        for (int d = 0; d < HD_; ++d) vT[d * 264 + n] = vs[d];
    }
    wait_async0();
    __syncthreads();

    // --- scores 64x256: wave (wm,wn): 16 rows x 128 cols = 8 WMMA tiles ---
    {
        int wm = wave & 3, wn = wave >> 2;
        v8f acc[8];
        for (int j = 0; j < 8; ++j) acc[j] = zero8();
        for (int ks = 0; ks < HD_; ks += 32) {
            v16bf a = frag_ld_a(qT, wm * 16 + (lane & 15), 72, ks, lane);
            for (int j = 0; j < 8; ++j) {
                v16bf bfr = frag_ld_b(kT, wn * 128 + j * 16 + (lane & 15), 72, ks, lane);
                acc[j] = wmma_bf16(a, bfr, acc[j]);
            }
        }
        int half8 = ((lane >> 4) & 1) * 8;
        for (int j = 0; j < 8; ++j) {
            int col = wn * 128 + j * 16 + (lane & 15);
            for (int r = 0; r < 8; ++r) {
                int rl = wm * 16 + r + half8;
                int rg = rowblk * 64 + rl;
                float s = acc[j][r] * 0.125f + pos[((size_t)h * N_ + rg) * N_ + col];
                int mv = mask[(rowbase + rg) * N_ + col];
                sS[rl * 256 + col] = (mv > 0) ? s : -1.0e9f;
            }
        }
    }
    __syncthreads();

    // --- softmax over 256 cols; 4 threads per row ---
    {
        int row = tid >> 2, q = tid & 3;
        float* sr = sS + row * 256 + q * 64;
        float mx = -3.0e38f;
        for (int j = 0; j < 64; ++j) mx = fmaxf(mx, sr[j]);
        red[row * 4 + q] = mx;
        __syncthreads();
        float rm = fmaxf(fmaxf(red[row * 4 + 0], red[row * 4 + 1]),
                         fmaxf(red[row * 4 + 2], red[row * 4 + 3]));
        float sum = 0.f;
        for (int j = 0; j < 64; ++j) { float e = __expf(sr[j] - rm); sr[j] = e; sum += e; }
        red[256 + row * 4 + q] = sum;
        __syncthreads();
        float inv = 1.0f / (red[256 + row * 4 + 0] + red[256 + row * 4 + 1] +
                            red[256 + row * 4 + 2] + red[256 + row * 4 + 3]);
        __bf16* pr = pT + row * 264 + q * 64;
        for (int j = 0; j < 64; ++j) pr[j] = (__bf16)(sr[j] * inv);
    }
    __syncthreads();

    // --- out = P(64x256) @ V(256x64): 16 tiles, 2 per wave ---
    {
        int half8 = ((lane >> 4) & 1) * 8;
        for (int i = 0; i < 2; ++i) {
            int ti = wave * 2 + i, tm = ti >> 2, tn = ti & 3;
            v8f acc = zero8();
            for (int ks = 0; ks < N_; ks += 32) {
                v16bf a   = frag_ld_a(pT, tm * 16 + (lane & 15), 264, ks, lane);
                v16bf bf2 = frag_ld_b(vT, tn * 16 + (lane & 15), 264, ks, lane);
                acc = wmma_bf16(a, bf2, acc);
            }
            int cb = h * HD_ + tn * 16 + (lane & 15);
            for (int r = 0; r < 8; ++r) {
                int n = rowblk * 64 + tm * 16 + r + half8;
                outa[(rowbase + n) * C_ + cb] = (__bf16)acc[r];
            }
        }
    }
}

// ---------------------------------------------------------------------------
// launch
// ---------------------------------------------------------------------------
extern "C" void kernel_launch(void* const* d_in, const int* in_sizes, int n_in,
                              void* d_out, int out_size, void* d_ws, size_t ws_size,
                              hipStream_t stream) {
    const float* x      = (const float*)d_in[0];
    const int*   mask   = (const int*)  d_in[1];
    const float* pos    = (const float*)d_in[2];
    const float* n1w    = (const float*)d_in[3];
    const float* n1b    = (const float*)d_in[4];
    const float* qkv_w  = (const float*)d_in[5];
    const float* qkv_b  = (const float*)d_in[6];
    const float* proj_w = (const float*)d_in[7];
    const float* proj_b = (const float*)d_in[8];
    const float* n2w    = (const float*)d_in[9];
    const float* n2b    = (const float*)d_in[10];
    const float* mlp_w1 = (const float*)d_in[11];
    const float* mlp_b1 = (const float*)d_in[12];
    const float* mlp_w2 = (const float*)d_in[13];
    const float* mlp_b2 = (const float*)d_in[14];
    float* out = (float*)d_out;

    char* p = (char*)d_ws;
    __bf16* w_qkv  = (__bf16*)p; p += (size_t)3 * C_ * C_ * 2;
    __bf16* w_proj = (__bf16*)p; p += (size_t)C_ * C_ * 2;
    __bf16* w_m1   = (__bf16*)p; p += (size_t)MLP_ * C_ * 2;
    __bf16* w_m2   = (__bf16*)p; p += (size_t)C_ * MLP_ * 2;
    __bf16* buf_xn   = (__bf16*)p; p += (size_t)M_TOT * C_ * 2;      // xn1 then xn2
    __bf16* buf_attn = (__bf16*)p; p += (size_t)M_TOT * C_ * 2;      // attn output
    float*  buf_x1   = (float*) p; p += (size_t)M_TOT * C_ * 4;      // x + attn (fp32)
    __bf16* buf_qkvh = (__bf16*)p;                                   // qkv, later mlp hidden

    // 1) weights -> bf16
    f32_to_bf16_kernel<<<(3 * C_ * C_ + 255) / 256, 256, 0, stream>>>(qkv_w,  w_qkv,  3 * C_ * C_);
    f32_to_bf16_kernel<<<(C_ * C_     + 255) / 256, 256, 0, stream>>>(proj_w, w_proj, C_ * C_);
    f32_to_bf16_kernel<<<(MLP_ * C_   + 255) / 256, 256, 0, stream>>>(mlp_w1, w_m1,   MLP_ * C_);
    f32_to_bf16_kernel<<<(C_ * MLP_   + 255) / 256, 256, 0, stream>>>(mlp_w2, w_m2,   C_ * MLP_);

    // 2) norm1(x) -> bf16
    groupnorm_kernel<<<B_ * P_, 256, 0, stream>>>(x, n1w, n1b, buf_xn);

    // 3) QKV GEMM: (32768x512) @ (1536x512)^T -> bf16 (32768x1536)
    gemm_bf16_kernel<0, true><<<dim3((3 * C_) / 128, M_TOT / 128), 256, 0, stream>>>(
        buf_xn, w_qkv, qkv_b, nullptr, buf_qkvh, 3 * C_, C_);

    // 4) attention per (b,p,h, 64-row strip)
    attn_kernel<<<dim3(N_ / 64, H_, B_ * P_), 256, ATTN_SMEM_BYTES, stream>>>(
        buf_qkvh, pos, mask, buf_attn);

    // 5) proj + residual(x) -> fp32 x1
    gemm_bf16_kernel<2, false><<<dim3(C_ / 128, M_TOT / 128), 256, 0, stream>>>(
        buf_attn, w_proj, proj_b, x, buf_x1, C_, C_);

    // 6) norm2(x1) -> bf16 (reuse buf_xn)
    groupnorm_kernel<<<B_ * P_, 256, 0, stream>>>(buf_x1, n2w, n2b, buf_xn);

    // 7) MLP up + GELU -> bf16 hidden (reuse buf_qkvh)
    gemm_bf16_kernel<1, true><<<dim3(MLP_ / 128, M_TOT / 128), 256, 0, stream>>>(
        buf_xn, w_m1, mlp_b1, nullptr, buf_qkvh, MLP_, C_);

    // 8) MLP down + residual(x1) -> fp32 output
    gemm_bf16_kernel<2, false><<<dim3(C_ / 128, M_TOT / 128), 256, 0, stream>>>(
        buf_qkvh, w_m2, mlp_b2, buf_x1, out, C_, MLP_);
}
